// KVCache_54116587930174
// MI455X (gfx1250) — compile-verified, matrix operand order checked by reading, not run on hardware
//
#include <hip/hip_runtime.h>
#include <hip/hip_bf16.h>

// ---------------------------------------------------------------------------
// MI455X (gfx1250) KV-cache attention, bf16 WMMA path + async global->LDS.
//   d_out = [ out (2*1024*2048) | K (2*16*4096*128) | V (2*16*4096*128) ] fp32
//   d_ws layout (56 MiB used):
//     0      : xb   bf16 x            (8 MiB)
//     8 MiB  : Wqb  bf16 Wq           (8 MiB)
//     16 MiB : Wkb  bf16 Wk           (8 MiB)
//     24 MiB : Wvb  bf16 Wv           (8 MiB)
//     32 MiB : Wob  bf16 Wo           (8 MiB)
//     40 MiB : Qb   bf16 Q [B,H,S,dk] (8 MiB)
//     48 MiB : Ab   bf16 attn-out [B,S,D] (8 MiB)
// ---------------------------------------------------------------------------

typedef __attribute__((ext_vector_type(16))) __bf16 v16bf;
typedef __attribute__((ext_vector_type(8)))  __bf16 v8bf;
typedef __attribute__((ext_vector_type(8)))  float  v8f;
typedef __attribute__((ext_vector_type(4)))  int    v4i;

#define D_MODEL 2048
#define NHEADS  16
#define DK      128
#define BATCH   2
#define SEQ     1024
#define PAST    3072
#define LTOT    4096   // PAST + SEQ

#if __has_builtin(__builtin_amdgcn_global_load_async_to_lds_b128) && \
    __has_builtin(__builtin_amdgcn_s_wait_asynccnt)
#define USE_ASYNC_LDS 1
#else
#define USE_ASYNC_LDS 0
#endif

__device__ __forceinline__ __bf16 f2bf(float f) {
  union { float f; unsigned u; } c{f};
  unsigned r = c.u + 0x7FFFu + ((c.u >> 16) & 1u);   // round-to-nearest-even
  unsigned short h = (unsigned short)(r >> 16);
  union { unsigned short s; __bf16 b; } o{h};
  return o.b;
}

__device__ __forceinline__ void lds_wave_fence() {
#if __has_builtin(__builtin_amdgcn_s_wait_dscnt)
  __builtin_amdgcn_s_wait_dscnt(0);
#else
  asm volatile("s_wait_dscnt 0" ::: "memory");
#endif
}

#if USE_ASYNC_LDS
// async DMA: 16 bytes global -> LDS, tracked by ASYNCcnt (bypasses VGPRs).
// Builtin signature (from hipcc diagnostic): param 0 is AS(1) v4i*, so cast
// accordingly; param 1 is the LDS destination, then imm offset + cpol.
__device__ __forceinline__ void async_b128(const void* g, void* l) {
  __builtin_amdgcn_global_load_async_to_lds_b128(
      (__attribute__((address_space(1))) v4i*)g,
      (__attribute__((address_space(3))) v4i*)l, 0, 0);
}
#endif

__device__ __forceinline__ v8f wmma_bf16(v16bf a, v16bf b, v8f c) {
  return __builtin_amdgcn_wmma_f32_16x16x32_bf16(
      /*neg_a=*/false, a, /*neg_b=*/false, b,
      /*c_mod=*/(short)0, c, /*reuse_a=*/false, /*reuse_b=*/false);
}

// A/B fragment where the per-lane dimension (M for A, N for B) is the SLOW
// axis of the source: elem(r, k) at base[r*stride + k].
// CDNA5 16-bit 16x32 layout: lane holds row (lane&15); VGPR v holds K pair
//   v<4  -> K = 2v, 2v+1   (+8 for lanes 16..31)
//   v>=4 -> K = 16+2(v-4)  (+8 for lanes 16..31)
template <typename PT>
__device__ __forceinline__ v16bf frag_rowmajor(PT base, int rowbase, int stride, int lane) {
  int row = rowbase + (lane & 15);
  int hi  = (lane & 16) ? 8 : 0;
  v16bf f;
#pragma unroll
  for (int v = 0; v < 8; ++v) {
    int kb = ((v < 4) ? (2 * v) : (16 + 2 * (v - 4))) + hi;
    f[2 * v]     = base[row * stride + kb];
    f[2 * v + 1] = base[row * stride + kb + 1];
  }
  return f;
}

// B fragment where the K dimension is the SLOW axis: elem(k, n) at
// base[k*stride + n]  (used for V tiles stored [key][dk]).
template <typename PT>
__device__ __forceinline__ v16bf frag_kmajor(PT base, int colbase, int stride, int lane) {
  int col = colbase + (lane & 15);
  int hi  = (lane & 16) ? 8 : 0;
  v16bf f;
#pragma unroll
  for (int v = 0; v < 8; ++v) {
    int kb = ((v < 4) ? (2 * v) : (16 + 2 * (v - 4))) + hi;
    f[2 * v]     = base[kb * stride + col];
    f[2 * v + 1] = base[(kb + 1) * stride + col];
  }
  return f;
}

// ---------------------------------------------------------------------------
// fp32 -> bf16 bulk convert (float4 vectorized)
// ---------------------------------------------------------------------------
__global__ void f32_to_bf16(const float* __restrict__ src, __bf16* __restrict__ dst, int n4) {
  int i = blockIdx.x * blockDim.x + threadIdx.x;
  if (i < n4) {
    float4 v = ((const float4*)src)[i];
    __bf16* d = dst + (size_t)i * 4;
    d[0] = f2bf(v.x); d[1] = f2bf(v.y); d[2] = f2bf(v.z); d[3] = f2bf(v.w);
  }
}

// ---------------------------------------------------------------------------
// past K/V -> output cache slots [B,H, 0:PAST, DK]   (float4 copy)
// ---------------------------------------------------------------------------
__global__ void copy_past(const float* __restrict__ pk, const float* __restrict__ pv,
                          float* __restrict__ Kc, float* __restrict__ Vc) {
  size_t i = (size_t)blockIdx.x * blockDim.x + threadIdx.x;   // float4 granularity
  const size_t per = (size_t)PAST * (DK / 4);                 // groups per (b,h)
  size_t bh = i / per, r = i % per;
  size_t t = r >> 5, dg = r & 31;                             // DK/4 == 32
  size_t o = (bh * LTOT + t) * (DK / 4) + dg;
  ((float4*)Kc)[o] = ((const float4*)pk)[i];
  ((float4*)Vc)[o] = ((const float4*)pv)[i];
}

// ---------------------------------------------------------------------------
// GEMM  Y[m][n] = sum_k A[m][k] * W[n][k] + bias[n]
//   block tile 128x128, BK=32, 8 waves, wave tile 64x32 (4x2 WMMA frags)
//   double-buffered LDS; async global->LDS staging overlapped with WMMA.
//   mode 0: Y -> bf16 outB in [B,H,S,dk]         (Q)
//   mode 1: Y -> fp32 outF cache layout t>=PAST  (K or V)
//   mode 2: Y -> fp32 outF [M,N]                 (final projection)
// ---------------------------------------------------------------------------
#define BM 128
#define BN 128
#define BK 32

__device__ __forceinline__ void stage_tile(const __bf16* __restrict__ ga,
                                           const __bf16* __restrict__ gw,
                                           __bf16* la, __bf16* lw) {
#if USE_ASYNC_LDS
  async_b128(ga,     la);
  async_b128(ga + 8, la + 8);
  async_b128(gw,     lw);
  async_b128(gw + 8, lw + 8);
#else
  v8bf a0 = *(const v8bf*)ga, a1 = *(const v8bf*)(ga + 8);
  v8bf w0 = *(const v8bf*)gw, w1 = *(const v8bf*)(gw + 8);
  *(v8bf*)la       = a0;
  *(v8bf*)(la + 8) = a1;
  *(v8bf*)lw       = w0;
  *(v8bf*)(lw + 8) = w1;
#endif
}

__global__ void __launch_bounds__(256)
gemm_bf16_wmma(const __bf16* __restrict__ A, const __bf16* __restrict__ W,
               const float* __restrict__ bias, int K, int N,
               int mode, float* __restrict__ outF, __bf16* __restrict__ outB) {
  __shared__ __attribute__((aligned(32))) __bf16 At[2][BM * BK];
  __shared__ __attribute__((aligned(32))) __bf16 Wt[2][BN * BK];

  const int tid  = threadIdx.x;
  const int lane = tid & 31;
  const int wave = tid >> 5;
  const int bm = blockIdx.y * BM;
  const int bn = blockIdx.x * BN;
  const int wm = (wave >> 2) * 64;   // 0 / 64
  const int wn = (wave & 3) * 32;    // 0 / 32 / 64 / 96

  v8f acc[4][2];
#pragma unroll
  for (int i = 0; i < 4; ++i)
#pragma unroll
    for (int j = 0; j < 2; ++j)
#pragma unroll
      for (int r = 0; r < 8; ++r) acc[i][j][r] = 0.0f;

  const int srow = tid >> 1;            // 0..127
  const int scol = (tid & 1) << 4;      // 0 / 16
  const __bf16* gA = A + (size_t)(bm + srow) * K + scol;
  const __bf16* gW = W + (size_t)(bn + srow) * K + scol;
  __bf16* lA0 = &At[0][srow * BK + scol];
  __bf16* lW0 = &Wt[0][srow * BK + scol];
  __bf16* lA1 = &At[1][srow * BK + scol];
  __bf16* lW1 = &Wt[1][srow * BK + scol];

  // prologue: stage k0 = 0 into buffer 0
  stage_tile(gA, gW, lA0, lW0);

  for (int k0 = 0, it = 0; k0 < K; k0 += BK, ++it) {
    const int cur = it & 1;
#if USE_ASYNC_LDS
    __builtin_amdgcn_s_wait_asynccnt(0);   // own DMA done; barrier makes all visible
#endif
    __syncthreads();

    // stage next K-step into the other buffer (overlaps with WMMAs below)
    if (k0 + BK < K)
      stage_tile(gA + (k0 + BK), gW + (k0 + BK),
                 cur ? lA0 : lA1, cur ? lW0 : lW1);

    const __bf16* Atc = At[cur];
    const __bf16* Wtc = Wt[cur];
    v16bf af[4], wf[2];
#pragma unroll
    for (int i = 0; i < 4; ++i) af[i] = frag_rowmajor(Atc, wm + i * 16, BK, lane);
#pragma unroll
    for (int j = 0; j < 2; ++j) wf[j] = frag_rowmajor(Wtc, wn + j * 16, BK, lane);
#pragma unroll
    for (int i = 0; i < 4; ++i)
#pragma unroll
      for (int j = 0; j < 2; ++j)
        acc[i][j] = wmma_bf16(af[i], wf[j], acc[i][j]);
  }

  // epilogue: C/D layout -> row = r + 8*(lane>=16), col = lane&15
  const int hi = (lane & 16) ? 8 : 0;
  const int cn = lane & 15;
#pragma unroll
  for (int i = 0; i < 4; ++i) {
#pragma unroll
    for (int j = 0; j < 2; ++j) {
#pragma unroll
      for (int r = 0; r < 8; ++r) {
        int gm = bm + wm + i * 16 + r + hi;      // b*SEQ + s
        int gn = bn + wn + j * 16 + cn;          // feature index
        float v = acc[i][j][r] + bias[gn];
        int bidx = gm >> 10, s = gm & (SEQ - 1);
        int h = gn >> 7, dk = gn & (DK - 1);
        if (mode == 0) {
          outB[((size_t)((bidx * NHEADS + h) * SEQ + s)) * DK + dk] = f2bf(v);
        } else if (mode == 1) {
          outF[((size_t)((bidx * NHEADS + h) * LTOT + PAST + s)) * DK + dk] = v;
        } else {
          outF[(size_t)gm * N + gn] = v;
        }
      }
    }
  }
}

// ---------------------------------------------------------------------------
// Flash attention: one block per (b,h, 128-row q tile); 8 waves x 16 q rows.
// Keys processed in chunks of 32; K/V staged fp32->bf16 into LDS by the block
// (conversion forces the VGPR path; next chunk is prefetched into cache).
// Per chunk per wave: 8 score WMMAs (Q 16x128 x K^T 128x32), online softmax,
// P transposed through per-wave LDS tile, 8 PV WMMAs (P 16x32 x V 32x128).
// ---------------------------------------------------------------------------
__global__ void __launch_bounds__(256)
attn_wmma(const __bf16* __restrict__ Qb, const float* __restrict__ Kc,
          const float* __restrict__ Vc, __bf16* __restrict__ Ab) {
  __shared__ __attribute__((aligned(32))) __bf16 Kt[32 * DK];
  __shared__ __attribute__((aligned(32))) __bf16 Vt[32 * DK];
  __shared__ __attribute__((aligned(32))) __bf16 Pt[8][16 * 32];

  const int tid  = threadIdx.x;
  const int lane = tid & 31;
  const int wave = tid >> 5;
  const int qt = blockIdx.x & 7;
  const int bh = blockIdx.x >> 3;          // b*NHEADS + h
  const int b  = bh >> 4;
  const int h  = bh & (NHEADS - 1);
  const int q0 = qt * 128 + wave * 16;     // wave's first q row

  // Q fragments (16 rows x 128 dk = 4 fragments of K=32), kept resident
  const __bf16* Qbase = Qb + (size_t)bh * SEQ * DK;
  v16bf qf[4];
#pragma unroll
  for (int kk = 0; kk < 4; ++kk)
    qf[kk] = frag_rowmajor(Qbase + kk * 32, q0, DK, lane);

  float rmax[8], rsum[8];
  v8f oacc[8];
#pragma unroll
  for (int r = 0; r < 8; ++r) { rmax[r] = -1e30f; rsum[r] = 0.0f; }
#pragma unroll
  for (int j = 0; j < 8; ++j)
#pragma unroll
    for (int r = 0; r < 8; ++r) oacc[j][r] = 0.0f;

  const float scale = 0.08838834764831845f;   // 1/sqrt(128)
  const float* Kg = Kc + (size_t)bh * LTOT * DK;
  const float* Vg = Vc + (size_t)bh * LTOT * DK;

  const int lrow = tid >> 3;          // 0..31 (key within chunk)
  const int lcol = (tid & 7) << 4;    // 0,16,...,112

  for (int c = 0; c < LTOT; c += 32) {
    // ---- stage K/V chunk (32 x 128) fp32 -> bf16 into LDS ----
    {
      const float4* kg = (const float4*)(Kg + (size_t)(c + lrow) * DK + lcol);
      const float4* vg = (const float4*)(Vg + (size_t)(c + lrow) * DK + lcol);
      if (c + 32 < LTOT) {  // pull next chunk toward L2/L0 while we compute
        __builtin_prefetch(Kg + (size_t)(c + 32 + lrow) * DK + lcol, 0, 3);
        __builtin_prefetch(Vg + (size_t)(c + 32 + lrow) * DK + lcol, 0, 3);
      }
#pragma unroll
      for (int e = 0; e < 4; ++e) {
        float4 kv = kg[e], vv = vg[e];
        int o = lrow * DK + lcol + e * 4;
        Kt[o + 0] = f2bf(kv.x); Kt[o + 1] = f2bf(kv.y);
        Kt[o + 2] = f2bf(kv.z); Kt[o + 3] = f2bf(kv.w);
        Vt[o + 0] = f2bf(vv.x); Vt[o + 1] = f2bf(vv.y);
        Vt[o + 2] = f2bf(vv.z); Vt[o + 3] = f2bf(vv.w);
      }
    }
    __syncthreads();

    // ---- scores: two 16x16 tiles over keys c..c+31 ----
    v8f s0, s1;
#pragma unroll
    for (int r = 0; r < 8; ++r) { s0[r] = 0.0f; s1[r] = 0.0f; }
#pragma unroll
    for (int kk = 0; kk < 4; ++kk) {
      s0 = wmma_bf16(qf[kk], frag_rowmajor(Kt + kk * 32, 0,  DK, lane), s0);
      s1 = wmma_bf16(qf[kk], frag_rowmajor(Kt + kk * 32, 16, DK, lane), s1);
    }

    float cm[8];
#pragma unroll
    for (int r = 0; r < 8; ++r) {
      s0[r] *= scale; s1[r] *= scale;
      cm[r] = fmaxf(s0[r], s1[r]);
    }
    // row-max across the 16 lanes holding each row (half-wave butterfly)
#pragma unroll
    for (int m = 1; m < 16; m <<= 1)
#pragma unroll
      for (int r = 0; r < 8; ++r) cm[r] = fmaxf(cm[r], __shfl_xor(cm[r], m));

    float alpha[8], p0[8], p1[8];
#pragma unroll
    for (int r = 0; r < 8; ++r) {
      float mn = fmaxf(rmax[r], cm[r]);
      alpha[r] = __expf(rmax[r] - mn);
      rmax[r] = mn;
      p0[r] = __expf(s0[r] - mn);
      p1[r] = __expf(s1[r] - mn);
      rsum[r] = alpha[r] * rsum[r] + p0[r] + p1[r];
    }
#pragma unroll
    for (int j = 0; j < 8; ++j)
#pragma unroll
      for (int r = 0; r < 8; ++r) oacc[j][r] *= alpha[r];

    // ---- transpose P (C layout -> A layout) through per-wave LDS tile ----
    {
      const int hi = (lane & 16) ? 8 : 0;
      const int cn = lane & 15;
      __bf16* pw = &Pt[wave][0];
#pragma unroll
      for (int r = 0; r < 8; ++r) {
        pw[(r + hi) * 32 + cn]      = f2bf(p0[r]);
        pw[(r + hi) * 32 + 16 + cn] = f2bf(p1[r]);
      }
    }
    lds_wave_fence();   // DS ops are in-order per wave; ensure load sees store

    v16bf pf = frag_rowmajor(&Pt[wave][0], 0, 32, lane);
#pragma unroll
    for (int j = 0; j < 8; ++j)
      oacc[j] = wmma_bf16(pf, frag_kmajor(Vt, j * 16, DK, lane), oacc[j]);

    __syncthreads();
  }

  // ---- finalize: reduce row sums, normalize, write bf16 [B,S,D] ----
#pragma unroll
  for (int m = 1; m < 16; m <<= 1)
#pragma unroll
    for (int r = 0; r < 8; ++r) rsum[r] += __shfl_xor(rsum[r], m);

  const int hi = (lane & 16) ? 8 : 0;
  const int cn = lane & 15;
#pragma unroll
  for (int j = 0; j < 8; ++j) {
#pragma unroll
    for (int r = 0; r < 8; ++r) {
      float v = oacc[j][r] / rsum[r];
      int s = q0 + r + hi;
      Ab[((size_t)(b * SEQ + s)) * D_MODEL + h * DK + j * 16 + cn] = f2bf(v);
    }
  }
}

// ---------------------------------------------------------------------------
extern "C" void kernel_launch(void* const* d_in, const int* in_sizes, int n_in,
                              void* d_out, int out_size, void* d_ws, size_t ws_size,
                              hipStream_t stream) {
  const float* x  = (const float*)d_in[0];
  const float* pk = (const float*)d_in[1];
  const float* pv = (const float*)d_in[2];
  const float* Wq = (const float*)d_in[3];
  const float* bq = (const float*)d_in[4];
  const float* Wk = (const float*)d_in[5];
  const float* bk = (const float*)d_in[6];
  const float* Wv = (const float*)d_in[7];
  const float* bv = (const float*)d_in[8];
  const float* Wo = (const float*)d_in[9];
  const float* bo = (const float*)d_in[10];

  float* out = (float*)d_out;                                  // [B,S,D]
  float* Kc  = out + (size_t)BATCH * SEQ * D_MODEL;            // [B,H,LTOT,DK]
  float* Vc  = Kc + (size_t)BATCH * NHEADS * LTOT * DK;

  char* ws = (char*)d_ws;
  const size_t MiB = 1024 * 1024;
  __bf16* xb  = (__bf16*)(ws + 0 * MiB);
  __bf16* Wqb = (__bf16*)(ws + 8 * MiB);
  __bf16* Wkb = (__bf16*)(ws + 16 * MiB);
  __bf16* Wvb = (__bf16*)(ws + 24 * MiB);
  __bf16* Wob = (__bf16*)(ws + 32 * MiB);
  __bf16* Qb  = (__bf16*)(ws + 40 * MiB);
  __bf16* Ab  = (__bf16*)(ws + 48 * MiB);

  const int n4 = (BATCH * SEQ * D_MODEL) / 4;                  // == D*D/4 too
  dim3 cb(256), cg(n4 / 256);
  f32_to_bf16<<<cg, cb, 0, stream>>>(x,  xb,  n4);
  f32_to_bf16<<<cg, cb, 0, stream>>>(Wq, Wqb, n4);
  f32_to_bf16<<<cg, cb, 0, stream>>>(Wk, Wkb, n4);
  f32_to_bf16<<<cg, cb, 0, stream>>>(Wv, Wvb, n4);
  f32_to_bf16<<<cg, cb, 0, stream>>>(Wo, Wob, n4);

  const size_t pastGroups = (size_t)BATCH * NHEADS * PAST * (DK / 4);
  copy_past<<<(unsigned)(pastGroups / 256), 256, 0, stream>>>(pk, pv, Kc, Vc);

  dim3 gg(D_MODEL / BN, (BATCH * SEQ) / BM);                   // 16 x 16
  gemm_bf16_wmma<<<gg, 256, 0, stream>>>(xb, Wqb, bq, D_MODEL, D_MODEL, 0, nullptr, Qb);
  gemm_bf16_wmma<<<gg, 256, 0, stream>>>(xb, Wkb, bk, D_MODEL, D_MODEL, 1, Kc, nullptr);
  gemm_bf16_wmma<<<gg, 256, 0, stream>>>(xb, Wvb, bv, D_MODEL, D_MODEL, 1, Vc, nullptr);

  attn_wmma<<<BATCH * NHEADS * (SEQ / 128), 256, 0, stream>>>(Qb, Kc, Vc, Ab);

  gemm_bf16_wmma<<<gg, 256, 0, stream>>>(Ab, Wob, bo, D_MODEL, D_MODEL, 2, out, nullptr);
}
